// RNN2_71073118814837
// MI455X (gfx1250) — matrix operand order repaired and checked by reading, not compiled, not run on hardware
//
#include <hip/hip_runtime.h>
#include <hip/hip_bf16.h>
#include <stdint.h>

// Problem dims (match reference)
#define T_STEPS 512
#define BATCH   64
#define DIM_IN  512
#define DIM_H   1024
#define DIM_OUT 128
#define CIN     (DIM_IN + DIM_H)   // 1536
#define C2      (2 * DIM_H)        // 2048

// Persistent-kernel geometry: 32 blocks x 8 waves = 256 waves,
// one 16x16 output tile per wave for each of the per-step GEMM tasks.
#define NBLK      32
#define NTHREADS  256
#define WPB       (NTHREADS / 32)

typedef __attribute__((ext_vector_type(16))) __bf16       v16bf;
typedef __attribute__((ext_vector_type(8)))  float        v8f;
typedef __attribute__((ext_vector_type(4)))  unsigned int v4u;

union Frag { v16bf f; v4u q[2]; };

// float -> bf16, round-to-nearest-even
__device__ __forceinline__ unsigned short f2bf(float x) {
  unsigned u = __float_as_uint(x);
  u += 0x7FFFu + ((u >> 16) & 1u);
  return (unsigned short)(u >> 16);
}

__global__ void cvt_f32_bf16(const float* __restrict__ src,
                             unsigned short* __restrict__ dst, int n) {
  int i = blockIdx.x * blockDim.x + threadIdx.x;
  int stride = gridDim.x * blockDim.x;
  for (; i < n; i += stride) dst[i] = f2bf(src[i]);
}

__global__ void init_barrier_k(unsigned* bar) { bar[0] = 0u; bar[1] = 0u; }

// Generation barrier: bar[0]=arrival counter, bar[1]=generation.
__device__ __forceinline__ void grid_barrier(unsigned* bar) {
  __threadfence();
  __syncthreads();
  if (threadIdx.x == 0) {
    unsigned* cnt = bar;
    unsigned* gen = bar + 1;
    unsigned g = __hip_atomic_load(gen, __ATOMIC_ACQUIRE, __HIP_MEMORY_SCOPE_AGENT);
    unsigned prev = __hip_atomic_fetch_add(cnt, 1u, __ATOMIC_ACQ_REL, __HIP_MEMORY_SCOPE_AGENT);
    if (prev == (unsigned)(NBLK - 1)) {
      __hip_atomic_store(cnt, 0u, __ATOMIC_RELAXED, __HIP_MEMORY_SCOPE_AGENT);
      __hip_atomic_fetch_add(gen, 1u, __ATOMIC_ACQ_REL, __HIP_MEMORY_SCOPE_AGENT);
    } else {
      while (__hip_atomic_load(gen, __ATOMIC_ACQUIRE, __HIP_MEMORY_SCOPE_AGENT) == g)
        __builtin_amdgcn_s_sleep(1);
    }
  }
  __syncthreads();
}

// A-fragment (16x32 bf16, M=lane&15, K split by lane-half): two 16B loads.
__device__ __forceinline__ Frag load_a(const unsigned short* row, int k0, int half) {
  Frag a;
  a.q[0] = *(const v4u*)(row + k0 + 8 * half);
  a.q[1] = *(const v4u*)(row + k0 + 16 + 8 * half);
  return a;
}
// B-fragment (32x16 bf16 = W.T tile; N=lane&15 picks weight row, K split by lane-half like A).
__device__ __forceinline__ Frag load_b(const unsigned short* wrow, int k0, int half) {
  Frag b;
  const unsigned short* p = wrow + k0 + 8 * half;
  b.q[0] = *(const v4u*)(p);
  b.q[1] = *(const v4u*)(p + 16);
  return b;
}

__device__ __forceinline__ v8f wmma_bf16(Frag a, Frag b, v8f c) {
  return __builtin_amdgcn_wmma_f32_16x16x32_bf16(
      false, a.f, false, b.f, (short)0, c, false, false);
}

__global__ void __launch_bounds__(NTHREADS, 1)
rnn_persistent(const unsigned short* __restrict__ xb,   // [T][B][IN] bf16
               const unsigned short* __restrict__ Wih,  // [H][CIN] bf16
               const unsigned short* __restrict__ Wiu,  // [H][CIN] bf16
               const unsigned short* __restrict__ Wuh,  // [H][C2]  bf16
               const unsigned short* __restrict__ Wuo,  // [OUT][C2] bf16
               const float* __restrict__ b_i2h,
               const float* __restrict__ b_i2u,
               const float* __restrict__ b_u2h,
               const float* __restrict__ b_u2o,
               unsigned short* __restrict__ ha,         // 2 x [B][H] ping-pong
               unsigned short* __restrict__ hb,         // 2 x [B][H] ping-pong
               unsigned short* __restrict__ ub,         // 2 x [B][H] ping-pong
               float* __restrict__ logits,              // [B][OUT]
               unsigned* __restrict__ bar) {
  const int lane = threadIdx.x & 31;
  const int wv   = threadIdx.x >> 5;
  const int gw   = blockIdx.x * WPB + wv;   // 0..255
  const int half = lane >> 4;
  const int l15  = lane & 15;

  // This wave's output tile: m0 = batch-tile (4 of), n0 = hidden-tile (64 of)
  const int m0 = (gw & 3) * 16;
  const int n0 = (gw >> 2) * 16;

  const float bh = b_i2h[n0 + l15];
  const float bu = b_i2u[n0 + l15];
  const float bb = b_u2h[n0 + l15];

  const unsigned short* w1row = Wih + (size_t)(n0 + l15) * CIN;
  const unsigned short* w2row = Wiu + (size_t)(n0 + l15) * CIN;
  const unsigned short* w3row = Wuh + (size_t)(n0 + l15) * C2;

  // Software pipeline: iteration t runs A(t) and B(t-1), ONE barrier per step.
  // A(t): new_ha(t), u(t)  from [x(t) | ha(t-1)]     (ha slot t&1 <- slot (t^1)&1)
  // B(s): new_hb(s)        from [u(s)  | hb(s-1)]    (hb slot s&1 <- slot (s^1)&1)
  // u double-buffered on t&1. B(T-1) is dead (output needs only u(T-1), hb(T-2)).
  for (int t = 0; t <= T_STEPS; ++t) {
    const int cur = t & 1;

    // ---------------- A(t): new_ha and u (shared A-fragment) ----------------
    if (t < T_STEPS) {
      const unsigned short* haP = ha + (cur ^ 1) * (BATCH * DIM_H);
      unsigned short*       haC = ha + cur * (BATCH * DIM_H);
      unsigned short*       uC  = ub + cur * (BATCH * DIM_H);

      v8f accH, accU;
#pragma unroll
      for (int j = 0; j < 8; ++j) { accH[j] = bh; accU[j] = bu; }

      const unsigned short* xrow  = xb  + ((size_t)t * BATCH + (m0 + l15)) * DIM_IN;
      const unsigned short* harow = haP + (size_t)(m0 + l15) * DIM_H;

      // K segment 1: x part (k = 0..511), no per-iteration source select
#pragma unroll 4
      for (int kk = 0; kk < DIM_IN / 32; ++kk) {
        const int k0 = kk * 32;
        Frag a  = load_a(xrow, k0, half);
        Frag b1 = load_b(w1row, k0, half);
        Frag b2 = load_b(w2row, k0, half);
        // stream next timestep's x rows toward the caches
        __builtin_prefetch((const void*)(xrow + (size_t)BATCH * DIM_IN + k0), 0, 1);
        accH = wmma_bf16(a, b1, accH);
        accU = wmma_bf16(a, b2, accU);
      }
      // K segment 2: ha part (k = 512..1535)
#pragma unroll 4
      for (int kk = 0; kk < DIM_H / 32; ++kk) {
        const int k0 = kk * 32;
        Frag a  = load_a(harow, k0, half);
        Frag b1 = load_b(w1row, DIM_IN + k0, half);
        Frag b2 = load_b(w2row, DIM_IN + k0, half);
        accH = wmma_bf16(a, b1, accH);
        accU = wmma_bf16(a, b2, accU);
      }
#pragma unroll
      for (int j = 0; j < 8; ++j) {
        const int m = m0 + j + 8 * half;           // C/D: VGPR j, lane-half -> M
        haC[(size_t)m * DIM_H + n0 + l15] = f2bf(accH[j]);
        uC [(size_t)m * DIM_H + n0 + l15] = f2bf(fmaxf(accU[j], 0.0f));
      }
    }

    // ---------------- B(t-1): new_hb (K over [u(t-1) | hb(t-2)]) ------------
    if (t >= 1 && t < T_STEPS) {           // B(T-1) result is never consumed
      const unsigned short* uP  = ub + (cur ^ 1) * (BATCH * DIM_H);
      const unsigned short* hbP = hb + cur * (BATCH * DIM_H);         // hb(t-2)
      unsigned short*       hbC = hb + (cur ^ 1) * (BATCH * DIM_H);   // hb(t-1)

      v8f accB;
#pragma unroll
      for (int j = 0; j < 8; ++j) accB[j] = bb;

      const unsigned short* urow  = uP  + (size_t)(m0 + l15) * DIM_H;
      const unsigned short* hbrow = hbP + (size_t)(m0 + l15) * DIM_H;

#pragma unroll 4
      for (int kk = 0; kk < DIM_H / 32; ++kk) {   // u part (k = 0..1023)
        const int k0 = kk * 32;
        accB = wmma_bf16(load_a(urow, k0, half), load_b(w3row, k0, half), accB);
      }
#pragma unroll 4
      for (int kk = 0; kk < DIM_H / 32; ++kk) {   // hb part (k = 1024..2047)
        const int k0 = kk * 32;
        accB = wmma_bf16(load_a(hbrow, k0, half),
                         load_b(w3row, DIM_H + k0, half), accB);
      }
#pragma unroll
      for (int j = 0; j < 8; ++j) {
        const int m = m0 + j + 8 * half;
        hbC[(size_t)m * DIM_H + n0 + l15] = f2bf(accB[j]);
      }
    }

    // -------- Final projection: logits = [u(T-1) | hb(T-2)] @ Wuo.T ---------
    if (t == T_STEPS && gw < 32) {
      const int mo = (gw & 3) * 16;
      const int no = (gw >> 2) * 16;     // 8 N-tiles cover OUT=128
      v8f accO;
      const float bo = b_u2o[no + l15];
#pragma unroll
      for (int j = 0; j < 8; ++j) accO[j] = bo;

      const unsigned short* urow  = ub + (cur ^ 1) * (BATCH * DIM_H)
                                  + (size_t)(mo + l15) * DIM_H;       // u(T-1)
      const unsigned short* hbrow = hb + cur * (BATCH * DIM_H)
                                  + (size_t)(mo + l15) * DIM_H;       // hb(T-2)
      const unsigned short* wrow  = Wuo + (size_t)(no + l15) * C2;

      for (int kk = 0; kk < DIM_H / 32; ++kk) {
        const int k0 = kk * 32;
        accO = wmma_bf16(load_a(urow, k0, half), load_b(wrow, k0, half), accO);
      }
      for (int kk = 0; kk < DIM_H / 32; ++kk) {
        const int k0 = kk * 32;
        accO = wmma_bf16(load_a(hbrow, k0, half),
                         load_b(wrow, DIM_H + k0, half), accO);
      }
#pragma unroll
      for (int j = 0; j < 8; ++j) {
        const int m = mo + j + 8 * half;
        logits[(size_t)m * DIM_OUT + no + l15] = accO[j];
      }
    }

    if (t < T_STEPS) grid_barrier(bar);   // no barrier needed after final proj
  }
}

__global__ void logsoftmax_k(const float* __restrict__ logits,
                             float* __restrict__ out) {
  __shared__ float red[DIM_OUT];
  const int b = blockIdx.x;
  const int i = threadIdx.x;
  float v = logits[b * DIM_OUT + i];
  red[i] = v;
  __syncthreads();
  for (int s = DIM_OUT / 2; s > 0; s >>= 1) {
    if (i < s) red[i] = fmaxf(red[i], red[i + s]);
    __syncthreads();
  }
  const float mx = red[0];
  __syncthreads();
  red[i] = expf(v - mx);
  __syncthreads();
  for (int s = DIM_OUT / 2; s > 0; s >>= 1) {
    if (i < s) red[i] += red[i + s];
    __syncthreads();
  }
  out[b * DIM_OUT + i] = v - mx - logf(red[0]);
}

// ---------------- workspace layout (bytes) ----------------
static constexpr size_t OFF_XB  = 0;
static constexpr size_t OFF_WIH = OFF_XB  + (size_t)T_STEPS * BATCH * DIM_IN * 2;
static constexpr size_t OFF_WIU = OFF_WIH + (size_t)DIM_H * CIN * 2;
static constexpr size_t OFF_WUH = OFF_WIU + (size_t)DIM_H * CIN * 2;
static constexpr size_t OFF_WUO = OFF_WUH + (size_t)DIM_H * C2 * 2;
static constexpr size_t OFF_HA  = OFF_WUO + (size_t)DIM_OUT * C2 * 2;
static constexpr size_t OFF_HB  = OFF_HA  + (size_t)2 * BATCH * DIM_H * 2;
static constexpr size_t OFF_U   = OFF_HB  + (size_t)2 * BATCH * DIM_H * 2;   // 2 slots
static constexpr size_t OFF_LOG = OFF_U   + (size_t)2 * BATCH * DIM_H * 2;
static constexpr size_t OFF_BAR = OFF_LOG + (size_t)BATCH * DIM_OUT * 4;
// total ~45.4 MB

extern "C" void kernel_launch(void* const* d_in, const int* in_sizes, int n_in,
                              void* d_out, int out_size, void* d_ws, size_t ws_size,
                              hipStream_t stream) {
  (void)in_sizes; (void)n_in; (void)out_size; (void)ws_size;
  const float* x     = (const float*)d_in[0];
  const float* h0    = (const float*)d_in[1];
  const float* W_i2h = (const float*)d_in[2];
  const float* b_i2h = (const float*)d_in[3];
  const float* W_i2u = (const float*)d_in[4];
  const float* b_i2u = (const float*)d_in[5];
  const float* W_u2h = (const float*)d_in[6];
  const float* b_u2h = (const float*)d_in[7];
  const float* W_u2o = (const float*)d_in[8];
  const float* b_u2o = (const float*)d_in[9];

  char* ws = (char*)d_ws;
  unsigned short* xb  = (unsigned short*)(ws + OFF_XB);
  unsigned short* wih = (unsigned short*)(ws + OFF_WIH);
  unsigned short* wiu = (unsigned short*)(ws + OFF_WIU);
  unsigned short* wuh = (unsigned short*)(ws + OFF_WUH);
  unsigned short* wuo = (unsigned short*)(ws + OFF_WUO);
  unsigned short* haW = (unsigned short*)(ws + OFF_HA);
  unsigned short* hbW = (unsigned short*)(ws + OFF_HB);
  unsigned short* uW  = (unsigned short*)(ws + OFF_U);
  float*          lg  = (float*)(ws + OFF_LOG);
  unsigned*       bar = (unsigned*)(ws + OFF_BAR);

  // bf16 staging of inputs (done every call: deterministic, inputs unchanged)
  cvt_f32_bf16<<<2048, 256, 0, stream>>>(x,     xb,  T_STEPS * BATCH * DIM_IN);
  cvt_f32_bf16<<<512,  256, 0, stream>>>(W_i2h, wih, DIM_H * CIN);
  cvt_f32_bf16<<<512,  256, 0, stream>>>(W_i2u, wiu, DIM_H * CIN);
  cvt_f32_bf16<<<512,  256, 0, stream>>>(W_u2h, wuh, DIM_H * C2);
  cvt_f32_bf16<<<128,  256, 0, stream>>>(W_u2o, wuo, DIM_OUT * C2);
  // h0 -> ping-pong slot 1 (prev slot for t=0 / s=0)
  cvt_f32_bf16<<<64, 256, 0, stream>>>(h0,                 haW + BATCH * DIM_H, BATCH * DIM_H);
  cvt_f32_bf16<<<64, 256, 0, stream>>>(h0 + BATCH * DIM_H, hbW + BATCH * DIM_H, BATCH * DIM_H);
  init_barrier_k<<<1, 1, 0, stream>>>(bar);

  rnn_persistent<<<NBLK, NTHREADS, 0, stream>>>(
      xb, wih, wiu, wuh, wuo, b_i2h, b_i2u, b_u2h, b_u2o,
      haW, hbW, uW, lg, bar);

  logsoftmax_k<<<BATCH, DIM_OUT, 0, stream>>>(lg, (float*)d_out);
}
